// KnowledgeEmbedding_77395310674420
// MI455X (gfx1250) — matrix operand order repaired and checked by reading, not compiled, not run on hardware
//
#include <hip/hip_runtime.h>
#include <hip/hip_bf16.h>
#include <math.h>

#define F_NUM 50000
#define H_NUM 10000
#define D_EMB 128
#define A_DIM 64
#define E_NUM 800000
#define B_NUM 1024
#define KC3   2000   // split-K chunk for final GEMM (25 chunks x 2000)
#define NTILES_TOT (E_NUM / 16)   // 50000 edge tiles
#define TILES_PER_BLK 32          // 512 edges per block

typedef __attribute__((ext_vector_type(16))) _Float16 v16h;
typedef __attribute__((ext_vector_type(8)))  _Float16 v8h;
typedef __attribute__((ext_vector_type(8)))  float    v8f;

// ---- fast transcendentals: use CDNA5 TRANS ops when the builtins exist
#if __has_builtin(__builtin_amdgcn_tanhf)
#define FAST_TANH(x) __builtin_amdgcn_tanhf(x)
#elif __has_builtin(__builtin_amdgcn_tanh_f32)
#define FAST_TANH(x) __builtin_amdgcn_tanh_f32(x)
#else
#define FAST_TANH(x) tanhf(x)
#endif

#if __has_builtin(__builtin_amdgcn_exp2f)
#define FAST_EXP(x) __builtin_amdgcn_exp2f((x) * 1.4426950408889634f)
#else
#define FAST_EXP(x) __expf(x)
#endif

#if __has_builtin(__builtin_amdgcn_rcpf)
#define FAST_RCP(x) __builtin_amdgcn_rcpf(x)
#else
#define FAST_RCP(x) (1.0f / (x))
#endif

__device__ __forceinline__ v16h cat8(v8h lo, v8h hi) {
  return __builtin_shufflevector(lo, hi, 0,1,2,3,4,5,6,7,8,9,10,11,12,13,14,15);
}

// ---------------------------------------------------------------- zero fill
__global__ void k_zero(float* __restrict__ p, long long n) {
  long long i = (long long)blockIdx.x * blockDim.x + threadIdx.x;
  long long stride = (long long)gridDim.x * blockDim.x;
  for (; i < n; i += stride) p[i] = 0.0f;
}

// ---------------------------------------------------------------- kernel 1
// Edge attention MLP: per 16-edge tile, GEMM [16x256]x[256x64] via WMMA f16,
// then bias + tanh + W_u dot + exp -> scores[e], atomic add into score_sum.
__global__ __launch_bounds__(128) void k_edge_att(
    const float* __restrict__ featEmb, const float* __restrict__ hidEmb,
    const float* __restrict__ Ww, const float* __restrict__ bw,
    const float* __restrict__ Wu, const int* __restrict__ connIdx,
    const int* __restrict__ connPart, float* __restrict__ scores,
    float* __restrict__ scoreSum)
{
  // W_w pre-swizzled into this kernel's per-lane B layout (contiguous k, same as A):
  // [kc(8)][nt(4)][lane(32)][j(16)] ; lane<16: N=lane+16*nt, K=kc*32+j
  //                                   lane>=16: N=lane-16+16*nt, K=kc*32+16+j
  __shared__ __align__(32) _Float16 sWwB[8 * 4 * 32 * 16];   // 32 KB
  __shared__ __align__(16) _Float16 sAtt[16][256];           // 8 KB
  __shared__ float sBias[A_DIM];
  __shared__ float sWu[A_DIM];
  __shared__ float sSc[16];

  const int tid = threadIdx.x;

  // coalesced walk over Ww (g = K*64 + N), scatter into the swizzled layout
  for (int g = tid; g < 2 * D_EMB * A_DIM; g += 128) {
    int K = g >> 6;            // 0..255
    int N = g & 63;            // 0..63
    int kc = K >> 5;
    int hs = (K >> 4) & 1;     // K half within 32
    int j  = K & 15;
    int ln = (N & 15) + hs * 16;
    int nt = N >> 4;
    sWwB[(((kc * 4 + nt) * 32) + ln) * 16 + j] = (_Float16)Ww[g];
  }
  if (tid < A_DIM) { sBias[tid] = bw[tid]; sWu[tid] = Wu[tid]; }
  if (tid < 16) sSc[tid] = 0.0f;
  __syncthreads();

  const int lane   = tid & 31;
  const int nt     = tid >> 5;     // wave id: N-tile 0..3
  const int halfid = lane >> 4;
  const int lmod   = lane & 15;

  for (int t = 0; t < TILES_PER_BLK; ++t) {
    const int tileIdx = blockIdx.x * TILES_PER_BLK + t;
    if (tileIdx >= NTILES_TOT) break;          // block-uniform
    const int ebase = tileIdx * 16;

    // ---- stage att_in tile [16 edges][256 dims] as f16 in LDS (b128 traffic)
    {
      int r  = tid >> 3;            // edge within tile
      int d0 = (tid & 7) * 32;      // dim range start
      int e  = ebase + r;
      const float* src;
      int off;
      if (d0 < 128) {
        int p = connPart[e];
        src = featEmb + (long long)p * D_EMB;
        off = d0;
      } else {
        int ci = connIdx[e];
        src = (ci < F_NUM) ? (featEmb + (long long)ci * D_EMB)
                           : (hidEmb + (long long)(ci - F_NUM) * D_EMB);
        off = d0 - 128;
      }
      const float4* s4 = (const float4*)(src + off);
      #pragma unroll
      for (int j = 0; j < 4; ++j) {
        float4 x = s4[2 * j];
        float4 y = s4[2 * j + 1];
        v8h h = { (_Float16)x.x, (_Float16)x.y, (_Float16)x.z, (_Float16)x.w,
                  (_Float16)y.x, (_Float16)y.y, (_Float16)y.z, (_Float16)y.w };
        *(v8h*)&sAtt[r][d0 + j * 8] = h;
      }
    }
    __syncthreads();

    // ---- WMMA: hidden[16 x 16cols] for this wave's N-tile, K = 256
    v8f c = {};
    #pragma unroll
    for (int kc = 0; kc < 8; ++kc) {
      int kb = kc * 32 + halfid * 16;
      v8h a0 = *(const v8h*)&sAtt[lmod][kb];
      v8h a1 = *(const v8h*)&sAtt[lmod][kb + 8];
      v16h a = cat8(a0, a1);
      v16h b = *(const v16h*)&sWwB[(((kc * 4 + nt) * 32) + lane) * 16];
      c = __builtin_amdgcn_wmma_f32_16x16x32_f16(false, a, false, b,
                                                 (short)0, c, false, false);
    }

    // ---- epilogue: bias + tanh + W_u, reduce over N (16 lanes + 4 waves)
    const int n  = lmod + nt * 16;
    float bias = sBias[n];
    float wu   = sWu[n];
    float part[8];
    #pragma unroll
    for (int v = 0; v < 8; ++v) part[v] = FAST_TANH(c[v] + bias) * wu;
    #pragma unroll
    for (int off = 1; off < 16; off <<= 1) {
      #pragma unroll
      for (int v = 0; v < 8; ++v)
        part[v] += __shfl_xor(part[v], off, 16);
    }
    if (lmod == 0) {
      #pragma unroll
      for (int v = 0; v < 8; ++v)
        atomicAdd(&sSc[halfid * 8 + v], part[v]);   // M = halfid*8 + v
    }
    __syncthreads();
    if (tid < 16) {
      int e = ebase + tid;
      float s = FAST_EXP(sSc[tid]);
      scores[e] = s;
      atomicAdd(&scoreSum[connPart[e]], s);
      sSc[tid] = 0.0f;
    }
    __syncthreads();
  }
}

// ---------------------------------------------------------------- kernel 2
// weights[e] = scores[e]/score_sum[p[e]]; context[p] += w * all_emb[idx].
// partition is sorted -> run-length accumulate per thread (one dim each).
__global__ __launch_bounds__(128) void k_context(
    const float* __restrict__ featEmb, const float* __restrict__ hidEmb,
    const int* __restrict__ connIdx, const int* __restrict__ connPart,
    const float* __restrict__ scores, const float* __restrict__ scoreSum,
    float* __restrict__ context)
{
  const int d = threadIdx.x;           // embedding dim 0..127
  const int e0 = blockIdx.x * 512;
  const int e1 = (e0 + 512 < E_NUM) ? e0 + 512 : E_NUM;

  float acc = 0.0f;
  float rs  = 0.0f;
  int cur = -1;
  for (int e = e0; e < e1; ++e) {
    int p = connPart[e];
    if (p != cur) {
      if (cur >= 0) atomicAdd(&context[cur * D_EMB + d], acc);
      cur = p;
      acc = 0.0f;
      rs  = FAST_RCP(scoreSum[p]);    // one rcp per segment
    }
    int ci = connIdx[e];
    const float* emb = (ci < F_NUM) ? (featEmb + (long long)ci * D_EMB)
                                    : (hidEmb + (long long)(ci - F_NUM) * D_EMB);
    float w = scores[e] * rs;
    acc += w * emb[d];
  }
  if (cur >= 0) atomicAdd(&context[cur * D_EMB + d], acc);
}

// ---------------------------------------------------------------- kernel 3
// out[1024,128] = values[1024,50000] @ context[50000,128], f16 WMMA, split-K.
// Block: 128 threads = 4 waves = 4 adjacent M-tiles; grid = (16 mquads, 25 k).
// sBt is double-buffered: next chunk prefetched to registers before WMMA.
__global__ __launch_bounds__(128) void k_final_gemm(
    const float* __restrict__ values, const float* __restrict__ context,
    float* __restrict__ out)
{
  __shared__ __align__(16) _Float16 sBt[2][128][40];  // context^T chunks, padded

  const int tid    = threadIdx.x;
  const int lane   = tid & 31;
  const int wid    = tid >> 5;
  const int halfid = lane >> 4;
  const int lmod   = lane & 15;
  const int mtile  = blockIdx.x * 4 + wid;                      // 0..63
  const long long arow = (long long)(mtile * 16 + lmod) * F_NUM;
  const int kbase = blockIdx.y * KC3;
  const int kend  = (kbase + KC3 < F_NUM) ? kbase + KC3 : F_NUM;

  // staging map: thread covers row sr (k within chunk), 32 consecutive cols
  const int sr  = tid >> 2;          // 0..31
  const int sc0 = (tid & 3) * 32;    // 0,32,64,96

  v8f zero = {};
  v8f acc[8];
  #pragma unroll
  for (int i = 0; i < 8; ++i) acc[i] = zero;

  // ---- initial stage into buffer 0
  {
    int kk = kbase + sr;
    bool ok = kk < kend;
    const float* srcp = context + (long long)kk * D_EMB + sc0;
    #pragma unroll
    for (int i = 0; i < 32; ++i)
      sBt[0][sc0 + i][sr] = ok ? (_Float16)srcp[i] : (_Float16)0.0f;
  }
  __syncthreads();

  int parity = 0;
  for (int k0 = kbase; k0 < kend; k0 += 32, parity ^= 1) {
    const int kn = k0 + 32;
    const bool haveNext = kn < kend;

    // ---- prefetch next B chunk into registers (overlaps with WMMA below)
    _Float16 pf[32];
    if (haveNext) {
      int kk = kn + sr;
      bool ok = kk < kend;
      const float* srcp = context + (long long)kk * D_EMB + sc0;
      #pragma unroll
      for (int i = 0; i < 32; ++i)
        pf[i] = ok ? (_Float16)srcp[i] : (_Float16)0.0f;
    }

    // ---- A: 16x32 f16 tile of values (f32 -> f16 on the fly, guarded tail)
    v16h a;
    const int kA0 = k0 + halfid * 8;
    #pragma unroll
    for (int j = 0; j < 8; ++j) {
      int k1 = kA0 + j;
      int k2 = kA0 + 16 + j;
      float f1 = (k1 < kend) ? values[arow + k1] : 0.0f;
      float f2 = (k2 < kend) ? values[arow + k2] : 0.0f;
      a[j]     = (_Float16)f1;
      a[8 + j] = (_Float16)f2;
    }

    // ---- 8 N-tiles from current buffer, A reused
    #pragma unroll
    for (int nt = 0; nt < 8; ++nt) {
      int n  = nt * 16 + lmod;
      int kb = halfid * 8;
      v8h b0 = *(const v8h*)&sBt[parity][n][kb];
      v8h b1 = *(const v8h*)&sBt[parity][n][kb + 16];
      v16h b = cat8(b0, b1);
      acc[nt] = __builtin_amdgcn_wmma_f32_16x16x32_f16(false, a, false, b,
                                                       (short)0, acc[nt], false, false);
    }

    // ---- commit prefetched chunk to the other buffer
    if (haveNext) {
      #pragma unroll
      for (int i = 0; i < 32; ++i)
        sBt[parity ^ 1][sc0 + i][sr] = pf[i];
    }
    __syncthreads();
  }

  // ---- split-K accumulate into out (zero-initialized each launch)
  const int row0 = mtile * 16 + halfid * 8;
  #pragma unroll
  for (int nt = 0; nt < 8; ++nt) {
    int col = nt * 16 + lmod;
    #pragma unroll
    for (int v = 0; v < 8; ++v)
      atomicAdd(&out[(row0 + v) * D_EMB + col], acc[nt][v]);
  }
}

// ---------------------------------------------------------------- launcher
extern "C" void kernel_launch(void* const* d_in, const int* in_sizes, int n_in,
                              void* d_out, int out_size, void* d_ws, size_t ws_size,
                              hipStream_t stream)
{
  const float* values  = (const float*)d_in[0];   // [B, F]
  const float* featEmb = (const float*)d_in[1];   // [F, D]
  const float* hidEmb  = (const float*)d_in[2];   // [H, D]
  const float* Ww      = (const float*)d_in[3];   // [2D, A]
  const float* bw      = (const float*)d_in[4];   // [A]
  const float* Wu      = (const float*)d_in[5];   // [A, 1]
  const int* connIdx   = (const int*)d_in[6];     // [E]
  const int* connPart  = (const int*)d_in[7];     // [E] (sorted)
  float* out = (float*)d_out;                     // [B, D]

  // Workspace layout (floats): scores[E] | score_sum[F] | context[F*D] ~ 29 MB
  float* ws       = (float*)d_ws;
  float* scores   = ws;
  float* scoreSum = scores + E_NUM;
  float* context  = scoreSum + F_NUM;

  // zero score_sum + context (contiguous) and out
  k_zero<<<2048, 256, 0, stream>>>(scoreSum,
      (long long)F_NUM + (long long)F_NUM * D_EMB);
  k_zero<<<256, 256, 0, stream>>>(out, (long long)B_NUM * D_EMB);

  // edge attention MLP (WMMA): 1563 blocks x 512 edges
  int nblk1 = (NTILES_TOT + TILES_PER_BLK - 1) / TILES_PER_BLK;
  k_edge_att<<<nblk1, 128, 0, stream>>>(featEmb, hidEmb, Ww, bw, Wu,
                                        connIdx, connPart, scores, scoreSum);

  // segment softmax + weighted segment sum -> context
  k_context<<<(E_NUM + 511) / 512, 128, 0, stream>>>(featEmb, hidEmb, connIdx,
                                                     connPart, scores, scoreSum,
                                                     context);

  // final GEMM (WMMA, double-buffered, split-K)
  dim3 g3(16, 25);
  k_final_gemm<<<g3, 128, 0, stream>>>(values, context, out);
}
